// MultiHeadAttention_34084860461210
// MI455X (gfx1250) — matrix operand order repaired and checked by reading, not compiled, not run on hardware
//
#include <hip/hip_runtime.h>
#include <cstddef>

// ---------- problem constants (match reference) ----------
constexpr int Bn  = 4;
constexpr int Sn  = 2048;
constexpr int Dn  = 1024;
constexpr int Hn  = 16;
constexpr int DKn = 64;
constexpr int Mtot = Bn * Sn;          // 8192
constexpr float SCALE = 0.125f;        // 1/sqrt(64)

typedef __bf16 bf16_t;
typedef __attribute__((ext_vector_type(16))) __bf16 bf16x16;
typedef __attribute__((ext_vector_type(8)))  __bf16 bf16x8;
typedef __attribute__((ext_vector_type(8)))  float  f32x8;
typedef __attribute__((ext_vector_type(4)))  int    i32x4_t;

// ---------- bf16 helpers ----------
__device__ __forceinline__ bf16_t to_bf16(float f) {
  unsigned u = __builtin_bit_cast(unsigned, f);
  unsigned r = u + 0x7FFFu + ((u >> 16) & 1u);   // round-to-nearest-even
  unsigned short h = (unsigned short)(r >> 16);
  return __builtin_bit_cast(bf16_t, h);
}
__device__ __forceinline__ unsigned pack2_bf16(float a, float b) {
  return (unsigned)__builtin_bit_cast(unsigned short, to_bf16(a)) |
         ((unsigned)__builtin_bit_cast(unsigned short, to_bf16(b)) << 16);
}

// ---------- async global->LDS copy (CDNA5 GLOBAL_LOAD_ASYNC_TO_LDS_B128) ----------
#if __has_builtin(__builtin_amdgcn_global_load_async_to_lds_b128)
#define HAVE_ASYNC_LDS 1
typedef __attribute__((address_space(1))) i32x4_t* gptr128_t;   // int4 in global AS
typedef __attribute__((address_space(3))) i32x4_t* lptr128_t;   // int4 in LDS AS
__device__ __forceinline__ void async_copy_b128(const void* g, void* l) {
  // generic->AS1: numeric identity; generic->AS3: low 32 bits are the LDS offset (ISA 10.2)
  gptr128_t gp = (gptr128_t)(unsigned long long)(uintptr_t)g;
  lptr128_t lp = (lptr128_t)(unsigned)(uintptr_t)l;
  __builtin_amdgcn_global_load_async_to_lds_b128(gp, lp, 0, 0);
}
__device__ __forceinline__ void async_wait0() {
  asm volatile("s_wait_asynccnt 0x0" ::: "memory");
}
#else
#define HAVE_ASYNC_LDS 0
__device__ __forceinline__ void async_copy_b128(const void* g, void* l) {
  *(uint4*)l = *(const uint4*)g;
}
__device__ __forceinline__ void async_wait0() {}
#endif

// ---------- WMMA fragment loaders (ISA 7.12.2 layouts) ----------
// A (16x32, 16-bit): lane half hl=lane>>4; elems 0..7 = K hl*8+0..7, elems 8..15 = K 16+hl*8+0..7
__device__ __forceinline__ bf16x16 load_frag_a(const bf16_t* row, int lane) {
  int h = (lane >> 4) << 3;
  bf16x8 lo = *(const bf16x8*)(row + h);
  bf16x8 hi = *(const bf16x8*)(row + 16 + h);
  bf16x16 r;
#pragma unroll
  for (int i = 0; i < 8; ++i) { r[i] = lo[i]; r[i + 8] = hi[i]; }
  return r;
}
// B (32x16, 16-bit): lane = N column; elems 0..7 = K hl*8+0..7, elems 8..15 = K 16+hl*8+0..7.
// `col` points at row [n][k=0] of an LDS [N][K] tile.
__device__ __forceinline__ bf16x16 load_frag_b(const bf16_t* col, int lane) {
  int h = (lane >> 4) << 3;
  bf16x8 lo = *(const bf16x8*)(col + h);
  bf16x8 hi = *(const bf16x8*)(col + 16 + h);
  bf16x16 r;
#pragma unroll
  for (int i = 0; i < 8; ++i) { r[i] = lo[i]; r[i + 8] = hi[i]; }
  return r;
}

__device__ __forceinline__ f32x8 wmma_bf16(bf16x16 a, bf16x16 b, f32x8 c) {
  return __builtin_amdgcn_wmma_f32_16x16x32_bf16(false, a, false, b, (short)0, c, false, false);
}

// ---------- GEMM + bias:  C[8192,1024] = A[8192,1024] * W[1024,1024] + bias ----------
// Block tile 128x128, K-panel 64, 8 waves; wave w owns 16 rows x full 128 cols.
// OUT_MODE 0: bf16 -> [B,H,S,DK] workspace.  OUT_MODE 1: fp32 -> [M,N] (d_out).
template <int OUT_MODE, typename AT>
__global__ __launch_bounds__(256) void gemm_bias_kernel(
    const AT* __restrict__ A, const float* __restrict__ W,
    const float* __restrict__ bias, void* __restrict__ Out) {
  __shared__ __align__(16) bf16_t sA[128][72];   // [m][k], 144B rows
  __shared__ __align__(16) bf16_t sB[128][72];   // W panel transposed: [n][k]

  const int tid  = threadIdx.x;
  const int lane = tid & 31;
  const int w    = tid >> 5;
  const int m0   = blockIdx.y * 128;
  const int n0   = blockIdx.x * 128;
  const int nl   = lane & 15, hl = lane >> 4;

  f32x8 acc[8];
#pragma unroll
  for (int nt = 0; nt < 8; ++nt) acc[nt] = (f32x8){};

  for (int kt = 0; kt < Dn / 64; ++kt) {
    const int k0 = kt * 64;
    if (kt + 1 < Dn / 64)
      __builtin_prefetch(&A[(size_t)(m0 + (tid >> 1)) * Dn + k0 + 64], 0, 1);

    // ---- stage A panel 128x64 ----
    if constexpr (sizeof(AT) == 4) {
      // fp32 source: float4 loads, packed bf16x4 stores
#pragma unroll
      for (int i = tid; i < 128 * 16; i += 256) {
        int r = i >> 4, c4 = i & 15;
        float4 f = *(const float4*)&A[(size_t)(m0 + r) * Dn + k0 + c4 * 4];
        uint2 p; p.x = pack2_bf16(f.x, f.y); p.y = pack2_bf16(f.z, f.w);
        *(uint2*)&sA[r][c4 * 4] = p;
      }
    } else {
      // bf16 source: async 16B copies
#pragma unroll
      for (int i = tid; i < 128 * 8; i += 256) {
        int r = i >> 3, c8 = i & 7;
        async_copy_b128(&A[(size_t)(m0 + r) * Dn + k0 + c8 * 8], &sA[r][c8 * 8]);
      }
      async_wait0();
    }
    // ---- stage W panel 64x128, transposed into [n][k] ----
#pragma unroll
    for (int i = tid; i < 64 * 32; i += 256) {
      int k = i >> 5, n4 = i & 31;
      float4 f = *(const float4*)&W[(size_t)(k0 + k) * Dn + n0 + n4 * 4];
      sB[n4 * 4 + 0][k] = to_bf16(f.x);
      sB[n4 * 4 + 1][k] = to_bf16(f.y);
      sB[n4 * 4 + 2][k] = to_bf16(f.z);
      sB[n4 * 4 + 3][k] = to_bf16(f.w);
    }
    __syncthreads();

    const bf16_t* arow = &sA[w * 16 + nl][0];
    bf16x16 a0 = load_frag_a(arow, lane);
    bf16x16 a1 = load_frag_a(arow + 32, lane);
#pragma unroll
    for (int nt = 0; nt < 8; ++nt) {
      const bf16_t* bcol = &sB[nt * 16 + nl][0];
      bf16x16 b0 = load_frag_b(bcol, lane);
      acc[nt] = wmma_bf16(a0, b0, acc[nt]);
      bf16x16 b1 = load_frag_b(bcol + 32, lane);
      acc[nt] = wmma_bf16(a1, b1, acc[nt]);
    }
    __syncthreads();
  }

  // epilogue: C layout row m = r + 8*(lane>=16), col n = lane&15
#pragma unroll
  for (int nt = 0; nt < 8; ++nt) {
    const int gn = n0 + nt * 16 + nl;
    const float bv = bias[gn];
#pragma unroll
    for (int r = 0; r < 8; ++r) {
      const int gm = m0 + w * 16 + r + hl * 8;
      const float val = acc[nt][r] + bv;
      if constexpr (OUT_MODE == 0) {
        const int b  = gm >> 11;
        const int s  = gm & (Sn - 1);
        const int hh = gn >> 6;
        const int dk = gn & (DKn - 1);
        ((bf16_t*)Out)[(((size_t)(b * Hn + hh) * Sn + s) << 6) + dk] = to_bf16(val);
      } else {
        ((float*)Out)[(size_t)gm * Dn + gn] = val;
      }
    }
  }
}

// ---------- causal flash attention over bf16 [B,H,S,DK] q/k/v ----------
// 256 threads (8 waves); 128-query tile per block; 64-key tiles; async K/Q staging.
__global__ __launch_bounds__(256) void attn_kernel(
    const bf16_t* __restrict__ Qw, const bf16_t* __restrict__ Kw,
    const bf16_t* __restrict__ Vw, bf16_t* __restrict__ Ow) {
  __shared__ __align__(16) bf16_t sQ[128][72];       // [q][dk]
  __shared__ __align__(16) bf16_t sK[64][72];        // [key][dk]  (B-frag layout for Q*K^T)
  __shared__ __align__(16) bf16_t sV[64][72];        // [dk][key]  (B-frag layout for P*V)
  __shared__ __align__(16) bf16_t sP[8][16][72];     // per-wave P tile [q][key]

  const int tid  = threadIdx.x;
  const int lane = tid & 31;
  const int w    = tid >> 5;
  const int hl   = lane >> 4, nl = lane & 15;
  const int qt   = blockIdx.x;
  const int bh   = blockIdx.y;
  const int b    = bh >> 4, h = bh & 15;
  const int q0   = qt * 128;

  const bf16_t* Qb = Qw + (size_t)bh * Sn * DKn;
  const bf16_t* Kb = Kw + (size_t)bh * Sn * DKn;
  const bf16_t* Vb = Vw + (size_t)bh * Sn * DKn;

  // load Q tile once (async b128 copies; rows are 128B)
#pragma unroll
  for (int j = tid; j < 128 * 8; j += 256) {
    int r = j >> 3, c8 = j & 7;
    async_copy_b128(Qb + (size_t)(q0 + r) * DKn + c8 * 8, &sQ[r][c8 * 8]);
  }
  async_wait0();
  __syncthreads();

  bf16x16 aq0 = load_frag_a(&sQ[w * 16 + nl][0], lane);
  bf16x16 aq1 = load_frag_a(&sQ[w * 16 + nl][32], lane);

  f32x8 o[4];
#pragma unroll
  for (int dt = 0; dt < 4; ++dt) o[dt] = (f32x8){};
  float mrow[8], lrow[8];
#pragma unroll
  for (int r = 0; r < 8; ++r) { mrow[r] = -1e30f; lrow[r] = 0.0f; }

  const int qlo = q0 + w * 16;                 // first query row of this wave
  for (int kt = 0; kt <= 2 * qt + 1; ++kt) {
    const int k0 = kt * 64;
    // ---- stage K (async b128) and V transposed (scalar, coalesced reads) ----
#pragma unroll
    for (int j = tid; j < 64 * 8; j += 256) {
      int r = j >> 3, c8 = j & 7;
      async_copy_b128(Kb + (size_t)(k0 + r) * DKn + c8 * 8, &sK[r][c8 * 8]);
    }
#pragma unroll
    for (int i = tid; i < 64 * 64; i += 256) {
      int dk = i & 63, key = i >> 6;
      sV[dk][key] = Vb[(size_t)(k0 + key) * DKn + dk];
    }
    async_wait0();
    __syncthreads();

    const bool active = (k0 <= qlo + 15);      // uniform per wave
    if (active) {
      // scores S = Q * K^T (DK=64 -> two k=32 WMMA steps per 16x16 tile)
      f32x8 st[4];
#pragma unroll
      for (int nt = 0; nt < 4; ++nt) {
        f32x8 z = (f32x8){};
        const bf16_t* kcol = &sK[nt * 16 + nl][0];
        bf16x16 bk0 = load_frag_b(kcol, lane);
        z = wmma_bf16(aq0, bk0, z);
        bf16x16 bk1 = load_frag_b(kcol + 32, lane);
        z = wmma_bf16(aq1, bk1, z);
        st[nt] = z;
      }

      const bool needMask = (k0 + 63 > qlo);
      float rmax[8];
#pragma unroll
      for (int r = 0; r < 8; ++r) rmax[r] = -1e30f;
#pragma unroll
      for (int nt = 0; nt < 4; ++nt) {
        const int kg = k0 + nt * 16 + nl;
#pragma unroll
        for (int r = 0; r < 8; ++r) {
          float v = st[nt][r] * SCALE;
          if (needMask) {
            const int qg = qlo + r + hl * 8;
            if (kg > qg) v = -1e30f;
          }
          st[nt][r] = v;
          rmax[r] = fmaxf(rmax[r], v);
        }
      }
#pragma unroll
      for (int r = 0; r < 8; ++r)
        for (int off = 8; off; off >>= 1)
          rmax[r] = fmaxf(rmax[r], __shfl_xor(rmax[r], off, 32));

      float corr[8], rsum[8];
#pragma unroll
      for (int r = 0; r < 8; ++r) {
        const float mn = fmaxf(mrow[r], rmax[r]);
        corr[r] = __expf(mrow[r] - mn);
        mrow[r] = mn;
        rsum[r] = 0.0f;
      }
#pragma unroll
      for (int nt = 0; nt < 4; ++nt)
#pragma unroll
        for (int r = 0; r < 8; ++r) {
          const float p = __expf(st[nt][r] - mrow[r]);
          rsum[r] += p;
          sP[w][r + hl * 8][nt * 16 + nl] = to_bf16(p);
        }
#pragma unroll
      for (int r = 0; r < 8; ++r) {
        for (int off = 8; off; off >>= 1)
          rsum[r] += __shfl_xor(rsum[r], off, 32);
        lrow[r] = lrow[r] * corr[r] + rsum[r];
      }
#pragma unroll
      for (int dt = 0; dt < 4; ++dt)
#pragma unroll
        for (int r = 0; r < 8; ++r) o[dt][r] *= corr[r];
    }
    __syncthreads();   // sP visible before re-reading as A fragments

    if (active) {
      bf16x16 ap0 = load_frag_a(&sP[w][nl][0], lane);
      bf16x16 ap1 = load_frag_a(&sP[w][nl][32], lane);
#pragma unroll
      for (int dt = 0; dt < 4; ++dt) {
        const bf16_t* vcol = &sV[dt * 16 + nl][0];
        bf16x16 bv0 = load_frag_b(vcol, lane);
        o[dt] = wmma_bf16(ap0, bv0, o[dt]);
        bf16x16 bv1 = load_frag_b(vcol + 32, lane);
        o[dt] = wmma_bf16(ap1, bv1, o[dt]);
      }
    }
    __syncthreads();   // done with sK/sV/sP before restage
  }

  // normalize; write bf16 attention output in [B,S,D] order (feeds Wo GEMM)
#pragma unroll
  for (int dt = 0; dt < 4; ++dt)
#pragma unroll
    for (int r = 0; r < 8; ++r) {
      const int s = qlo + r + hl * 8;
      const float val = o[dt][r] / lrow[r];
      Ow[(size_t)(b * Sn + s) * Dn + h * DKn + dt * 16 + nl] = to_bf16(val);
    }
}

// ---------- host launcher ----------
extern "C" void kernel_launch(void* const* d_in, const int* in_sizes, int n_in,
                              void* d_out, int out_size, void* d_ws, size_t ws_size,
                              hipStream_t stream) {
  (void)in_sizes; (void)n_in; (void)out_size; (void)ws_size;
  const float* Q  = (const float*)d_in[0];
  const float* K  = (const float*)d_in[1];
  const float* V  = (const float*)d_in[2];
  const float* Wq = (const float*)d_in[3];
  const float* bq = (const float*)d_in[4];
  const float* Wk = (const float*)d_in[5];
  const float* bk = (const float*)d_in[6];
  const float* Wv = (const float*)d_in[7];
  const float* bv = (const float*)d_in[8];
  const float* Wo = (const float*)d_in[9];
  const float* bo = (const float*)d_in[10];
  // d_in[11] = causal mask (statically known -> ignored)

  const size_t perTensor = (size_t)Bn * Hn * Sn * DKn;
  bf16_t* qws = (bf16_t*)d_ws;
  bf16_t* kws = qws + perTensor;
  bf16_t* vws = kws + perTensor;
  bf16_t* aws = vws + perTensor;   // [B,S,D] bf16

  dim3 gg(Dn / 128, Mtot / 128);   // (8, 64)
  gemm_bias_kernel<0, float><<<gg, 256, 0, stream>>>(Q, Wq, bq, qws);
  gemm_bias_kernel<0, float><<<gg, 256, 0, stream>>>(K, Wk, bk, kws);
  gemm_bias_kernel<0, float><<<gg, 256, 0, stream>>>(V, Wv, bv, vws);

  attn_kernel<<<dim3(Sn / 128, Bn * Hn), 256, 0, stream>>>(qws, kws, vws, aws);

  gemm_bias_kernel<1, bf16_t><<<gg, 256, 0, stream>>>(aws, Wo, bo, (float*)d_out);
}
